// GAFR_31817117729113
// MI455X (gfx1250) — matrix operand order repaired and checked by reading, not compiled, not run on hardware
//
#include <hip/hip_runtime.h>

#define HEADS 8
#define B_    32
#define C_    312
#define N_    1024
#define D_    128
#define HW    1024     // 32*32
#define CP    384      // padded token count (mult of 128)
#define MROWS (B_*C_)  // 9984

typedef __bf16 bf16_t;
typedef __attribute__((ext_vector_type(16))) __bf16 v16bf;
typedef __attribute__((ext_vector_type(8)))  __bf16 v8bf;
typedef __attribute__((ext_vector_type(8)))  float  v8f;

union Frag { v16bf v; v8bf h[2]; };

__device__ __forceinline__ bf16_t f2bf(float f) {
    unsigned u = __builtin_bit_cast(unsigned, f);
    u += 0x7FFFu + ((u >> 16) & 1u);            // round-to-nearest-even
    unsigned short hs = (unsigned short)(u >> 16);
    return __builtin_bit_cast(bf16_t, hs);
}

__device__ __forceinline__ float hswish(float x) {
    float t = fminf(fmaxf(x + 3.0f, 0.0f), 6.0f);
    return x * t * (1.0f / 6.0f);
}

// ---------------------------------------------------------------- cast f32 -> bf16
__global__ void k_cast_bf16(const float* __restrict__ src, bf16_t* __restrict__ dst, int n) {
    int i = blockIdx.x * blockDim.x + threadIdx.x;
    if (i < n) dst[i] = f2bf(src[i]);
}

// ---------------------------------------------------------------- LayerNorm x3 (q/k/v) -> bf16
__global__ __launch_bounds__(256) void k_ln3(
    const float* __restrict__ x,
    const float* __restrict__ qg, const float* __restrict__ qb,
    const float* __restrict__ kg, const float* __restrict__ kb,
    const float* __restrict__ vg, const float* __restrict__ vb,
    bf16_t* __restrict__ xq, bf16_t* __restrict__ xk, bf16_t* __restrict__ xv)
{
    int r = blockIdx.x;
    const float* xr = x + (long)r * N_;
    float vals[4]; float s = 0.f, s2 = 0.f;
    #pragma unroll
    for (int i = 0; i < 4; ++i) {
        float v = xr[threadIdx.x + 256 * i];
        vals[i] = v; s += v; s2 += v * v;
    }
    __shared__ float r1[256], r2[256];
    r1[threadIdx.x] = s; r2[threadIdx.x] = s2; __syncthreads();
    for (int o = 128; o > 0; o >>= 1) {
        if (threadIdx.x < o) { r1[threadIdx.x] += r1[threadIdx.x + o]; r2[threadIdx.x] += r2[threadIdx.x + o]; }
        __syncthreads();
    }
    float mu  = r1[0] * (1.0f / N_);
    float var = r2[0] * (1.0f / N_) - mu * mu;
    float inv = rsqrtf(var + 1e-5f);
    #pragma unroll
    for (int i = 0; i < 4; ++i) {
        int j = threadIdx.x + 256 * i;
        float xn = (vals[i] - mu) * inv;
        long o = (long)r * N_ + j;
        xq[o] = f2bf(xn * qg[j] + qb[j]);
        xk[o] = f2bf(xn * kg[j] + kb[j]);
        xv[o] = f2bf(xn * vg[j] + vb[j]);
    }
}

// ---------------------------------------------------------------- WMMA GEMM helpers
// Fragment loaders follow the CDNA5 VGPR layouts:
//   A 16x32 bf16: lane L<16 holds row m0+L, K {k0..k0+7, k0+16..k0+23}; lanes 16..31 shift K by 8.
//   B 32x16 bf16 (from (N,K)-row-major operand): lane L<16 holds col n0+L, K {k0..k0+7, k0+16..k0+23}; hi lanes shift K by 8.
__device__ __forceinline__ void ld_frags(const bf16_t* __restrict__ ap0,
                                         const bf16_t* __restrict__ ap1,
                                         const bf16_t* __restrict__ bp, long bstride,
                                         int k0, Frag (&a)[2], Frag (&b)[4])
{
    a[0].h[0] = *(const v8bf*)(ap0 + k0); a[0].h[1] = *(const v8bf*)(ap0 + k0 + 16);
    a[1].h[0] = *(const v8bf*)(ap1 + k0); a[1].h[1] = *(const v8bf*)(ap1 + k0 + 16);
    #pragma unroll
    for (int t = 0; t < 4; ++t) {
        const bf16_t* q = bp + k0 + t * bstride;
        b[t].h[0] = *(const v8bf*)q;
        b[t].h[1] = *(const v8bf*)(q + 16);
    }
}

__device__ __forceinline__ void mma8(const Frag (&a)[2], const Frag (&b)[4], v8f (&acc)[2][4]) {
    #pragma unroll
    for (int i = 0; i < 2; ++i)
        #pragma unroll
        for (int t = 0; t < 4; ++t)
            acc[i][t] = __builtin_amdgcn_wmma_f32_16x16x32_bf16(
                false, a[i].v, false, b[t].v, (short)0, acc[i][t], false, false);
}

// C[m,n] = sum_k A[m,k] * Bw[n,k]  (both row-major, leading dim K), f32 accum.
// Block: 256 threads = 8 waves (4Mx2N); wave tile 32x64; block tile 128x128.
// K must be a multiple of 64 (true for 1024 / 384 / 128).
// Software-pipelined: fragments for step k+32 are in flight while step k's 8 WMMAs issue.
__global__ __launch_bounds__(256) void k_gemm_bf16(
    const bf16_t* __restrict__ A, const bf16_t* __restrict__ Bw, float* __restrict__ Cc,
    int K, int ldc, int Mv, long sA, long sB, long sC, const float* __restrict__ bias)
{
    long z = blockIdx.z;
    A  += z * sA; Bw += z * sB; Cc += z * sC;
    int wave = threadIdx.x >> 5;
    int lane = threadIdx.x & 31;
    int wm = wave >> 1, wn = wave & 1;
    int m0 = blockIdx.x * 128 + wm * 32;
    int n0 = blockIdx.y * 128 + wn * 64;
    int lrow = lane & 15;
    int hi   = (lane >> 4) & 1;

    const bf16_t* ap0 = A  + (long)(m0 + lrow) * K + hi * 8;
    const bf16_t* ap1 = ap0 + 16L * K;
    const bf16_t* bp  = Bw + (long)(n0 + lrow) * K + hi * 8;
    const long bstride = 16L * K;

    v8f acc[2][4] = {};
    Frag aX[2], bX[4], aY[2], bY[4];

    ld_frags(ap0, ap1, bp, bstride, 0, aX, bX);
    for (int k0 = 0; k0 < K - 64; k0 += 64) {
        ld_frags(ap0, ap1, bp, bstride, k0 + 32, aY, bY);
        mma8(aX, bX, acc);
        ld_frags(ap0, ap1, bp, bstride, k0 + 64, aX, bX);
        mma8(aY, bY, acc);
    }
    ld_frags(ap0, ap1, bp, bstride, K - 32, aY, bY);
    mma8(aX, bX, acc);
    mma8(aY, bY, acc);

    int col = n0 + lrow;
    #pragma unroll
    for (int i = 0; i < 2; ++i) {
        int rbase = m0 + i * 16 + hi * 8;
        #pragma unroll
        for (int t = 0; t < 4; ++t) {
            int c = col + t * 16;
            float bv = bias ? bias[c] : 0.0f;
            #pragma unroll
            for (int e = 0; e < 8; ++e) {
                int rr = rbase + e;
                if (rr < Mv) Cc[(long)rr * ldc + c] = acc[i][t][e] + bv;
            }
        }
    }
}

// ---------------------------------------------------------------- depthwise convs (groups 1..3)
__global__ void k_dw(const float* __restrict__ t,
                     const float* __restrict__ dw1, const float* __restrict__ dw2,
                     const float* __restrict__ dw0, float* __restrict__ out)
{
    long idx = (long)blockIdx.x * blockDim.x + threadIdx.x;
    if (idx >= (long)B_ * 234 * HW) return;
    int pix = idx & (HW - 1);
    int j   = (int)((idx >> 10) % 234);
    int b   = (int)(idx / ((long)234 * HW));
    int hh = pix >> 5, ww = pix & 31;
    const float* src = t + ((long)b * C_ + 78 + j) * HW;
    float acc = 0.f;
    int g = j / 78;
    if (g == 1) {                                        // 5x5, pad 2 (dw2)
        const float* w = dw2 + (j - 78) * 25;
        for (int ky = 0; ky < 5; ++ky) { int y = hh + ky - 2; if ((unsigned)y >= 32u) continue;
            for (int kx = 0; kx < 5; ++kx) { int xx = ww + kx - 2; if ((unsigned)xx >= 32u) continue;
                acc += w[ky * 5 + kx] * src[y * 32 + xx]; } }
    } else {                                             // 3x3, pad 1 (dw1 or dw0)
        const float* w = (g == 0) ? dw1 + j * 9 : dw0 + (j - 156) * 9;
        for (int ky = 0; ky < 3; ++ky) { int y = hh + ky - 1; if ((unsigned)y >= 32u) continue;
            for (int kx = 0; kx < 3; ++kx) { int xx = ww + kx - 1; if ((unsigned)xx >= 32u) continue;
                acc += w[ky * 3 + kx] * src[y * 32 + xx]; } }
    }
    out[idx] = acc;
}

// ---------------------------------------------------------------- pointwise 78x78 convs
__global__ void k_pw(const float* __restrict__ dwb,
                     const float* __restrict__ pw1, const float* __restrict__ pw2,
                     const float* __restrict__ pw0, float* __restrict__ out)
{
    long idx = (long)blockIdx.x * blockDim.x + threadIdx.x;
    if (idx >= (long)B_ * 234 * HW) return;
    int pix = idx & (HW - 1);
    int ocg = (int)((idx >> 10) % 234);
    int b   = (int)(idx / ((long)234 * HW));
    int g = ocg / 78, oc = ocg % 78;
    const float* w  = ((g == 0) ? pw1 : (g == 1) ? pw2 : pw0) + oc * 78;
    const float* in = dwb + ((long)b * 234 + g * 78) * HW + pix;
    float acc = 0.f;
    for (int ic = 0; ic < 78; ++ic) acc += w[ic] * in[(long)ic * HW];
    out[idx] = acc;
}

// ---------------------------------------------------------------- GroupNorm stats for group-3 block
__global__ __launch_bounds__(256) void k_gnstats(const float* __restrict__ pwb, float* __restrict__ st) {
    int b = blockIdx.x >> 1, half = blockIdx.x & 1;
    const float* base = pwb + ((long)b * 234 + 156 + half * 39) * HW;
    float s = 0.f, s2 = 0.f;
    for (int i = threadIdx.x; i < 39 * HW; i += 256) { float v = base[i]; s += v; s2 += v * v; }
    __shared__ float r1[256], r2[256];
    r1[threadIdx.x] = s; r2[threadIdx.x] = s2; __syncthreads();
    for (int o = 128; o > 0; o >>= 1) {
        if (threadIdx.x < o) { r1[threadIdx.x] += r1[threadIdx.x + o]; r2[threadIdx.x] += r2[threadIdx.x + o]; }
        __syncthreads();
    }
    if (threadIdx.x == 0) {
        float cnt = 39.0f * HW;
        float mu = r1[0] / cnt;
        float var = r2[0] / cnt - mu * mu;
        st[blockIdx.x * 2 + 0] = mu;
        st[blockIdx.x * 2 + 1] = rsqrtf(var + 1e-5f);
    }
}

// ---------------------------------------------------------------- BN/GN + hswish + head-scatter to bf16
__global__ void k_finalize(const float* __restrict__ t, const float* __restrict__ pwb,
    const float* __restrict__ gst,
    const float* __restrict__ bn0g, const float* __restrict__ bn0b, const float* __restrict__ bn0m, const float* __restrict__ bn0v,
    const float* __restrict__ bn1g, const float* __restrict__ bn1b, const float* __restrict__ bn1m, const float* __restrict__ bn1v,
    const float* __restrict__ bn2g, const float* __restrict__ bn2b, const float* __restrict__ bn2m, const float* __restrict__ bn2v,
    const float* __restrict__ gng, const float* __restrict__ gnb,
    bf16_t* __restrict__ out_std, bf16_t* __restrict__ out_tr)
{
    long idx = (long)blockIdx.x * blockDim.x + threadIdx.x;
    if (idx >= (long)B_ * C_ * HW) return;
    int pix = idx & (HW - 1);
    int c   = (int)((idx >> 10) % C_);
    int b   = (int)(idx / ((long)C_ * HW));
    float val;
    if (c < 78) {
        float x = t[idx];
        float sc = bn0g[c] * rsqrtf(bn0v[c] + 1e-5f);
        val = (x - bn0m[c]) * sc + bn0b[c];
    } else {
        float x = pwb[((long)b * 234 + (c - 78)) * HW + pix];
        if (c < 156)      { int ch = c - 78;  float sc = bn1g[ch] * rsqrtf(bn1v[ch] + 1e-5f); val = (x - bn1m[ch]) * sc + bn1b[ch]; }
        else if (c < 234) { int ch = c - 156; float sc = bn2g[ch] * rsqrtf(bn2v[ch] + 1e-5f); val = (x - bn2m[ch]) * sc + bn2b[ch]; }
        else {
            int ch = c - 234; int half = ch / 39;
            float mu = gst[(b * 2 + half) * 2 + 0], inv = gst[(b * 2 + half) * 2 + 1];
            val = (x - mu) * inv * gng[ch] + gnb[ch];
        }
    }
    val = hswish(val);
    int hd = pix >> 7, dd = pix & 127;
    long bh = (long)b * HEADS + hd;
    bf16_t bv = f2bf(val);
    if (out_std) out_std[(bh * CP + c) * D_ + dd] = bv;       // (BH, CP, d)
    if (out_tr)  out_tr [(bh * D_ + dd) * CP + c] = bv;       // (BH, d, CP)  (v transposed)
}

// ---------------------------------------------------------------- gelu * scale + softmax; write attn f32 + bf16
__global__ __launch_bounds__(128) void k_softmax(const float* __restrict__ scores,
                                                 float* __restrict__ attn_out,
                                                 bf16_t* __restrict__ attn_bf)
{
    int  c  = blockIdx.x;        // 0..CP-1
    long bh = blockIdx.y;        // 0..255
    bf16_t* abrow = attn_bf + (bh * CP + c) * CP;
    if (c >= C_) {               // pad rows: zero so GEMM3 stays finite (outputs ignored anyway)
        for (int k = threadIdx.x; k < CP; k += 128) abrow[k] = f2bf(0.0f);
        return;
    }
    const float* srow = scores + (bh * CP + c) * CP;
    const float SCALE = 0.08838834764831845f;   // d^-0.5, d=128
    float g[3]; float mx = -1e30f;
    #pragma unroll
    for (int i = 0; i < 3; ++i) {
        int k = threadIdx.x + i * 128;
        if (k < C_) {
            float s = srow[k];
            float gg = 0.5f * s * (1.0f + erff(s * 0.70710678118654752f)) * SCALE;
            g[i] = gg; mx = fmaxf(mx, gg);
        } else g[i] = -1e30f;
    }
    __shared__ float red[128];
    red[threadIdx.x] = mx; __syncthreads();
    for (int o = 64; o > 0; o >>= 1) { if (threadIdx.x < o) red[threadIdx.x] = fmaxf(red[threadIdx.x], red[threadIdx.x + o]); __syncthreads(); }
    mx = red[0]; __syncthreads();
    float e[3]; float sum = 0.f;
    #pragma unroll
    for (int i = 0; i < 3; ++i) {
        int k = threadIdx.x + i * 128;
        e[i] = (k < C_) ? __expf(g[i] - mx) : 0.0f;
        sum += e[i];
    }
    red[threadIdx.x] = sum; __syncthreads();
    for (int o = 64; o > 0; o >>= 1) { if (threadIdx.x < o) red[threadIdx.x] += red[threadIdx.x + o]; __syncthreads(); }
    float inv = 1.0f / red[0];
    float* arow = attn_out + (bh * C_ + c) * (long)C_;
    #pragma unroll
    for (int i = 0; i < 3; ++i) {
        int k = threadIdx.x + i * 128;
        float p = e[i] * inv;
        if (k < C_) { arow[k] = p; abrow[k] = f2bf(p); }
        else if (k < CP) abrow[k] = f2bf(0.0f);
    }
}

// ---------------------------------------------------------------- row L2-normalize (in place on d_out)
__global__ __launch_bounds__(256) void k_rownorm(float* __restrict__ out) {
    int r = blockIdx.x;
    float* row = out + (long)r * N_;
    float v[4]; float s = 0.f;
    #pragma unroll
    for (int i = 0; i < 4; ++i) { v[i] = row[threadIdx.x + 256 * i]; s += v[i] * v[i]; }
    __shared__ float red[256];
    red[threadIdx.x] = s; __syncthreads();
    for (int o = 128; o > 0; o >>= 1) { if (threadIdx.x < o) red[threadIdx.x] += red[threadIdx.x + o]; __syncthreads(); }
    float inv = 1.0f / fmaxf(sqrtf(red[0]), 1e-12f);
    #pragma unroll
    for (int i = 0; i < 4; ++i) row[threadIdx.x + 256 * i] = v[i] * inv;
}

// ================================================================ host orchestration
extern "C" void kernel_launch(void* const* d_in, const int* in_sizes, int n_in,
                              void* d_out, int out_size, void* d_ws, size_t ws_size,
                              hipStream_t stream)
{
    (void)in_sizes; (void)n_in; (void)out_size; (void)ws_size;
    const float* x      = (const float*)d_in[0];
    const float* ln_qg  = (const float*)d_in[3];  const float* ln_qb = (const float*)d_in[4];
    const float* ln_kg  = (const float*)d_in[5];  const float* ln_kb = (const float*)d_in[6];
    const float* ln_vg  = (const float*)d_in[7];  const float* ln_vb = (const float*)d_in[8];
    const float* wq_f   = (const float*)d_in[9];
    const float* wk_f   = (const float*)d_in[10];
    const float* wv_f   = (const float*)d_in[11];
    const float* wp_f   = (const float*)d_in[12];
    const float* b_proj = (const float*)d_in[13];
    const float* dw0    = (const float*)d_in[14]; const float* pw0 = (const float*)d_in[15];
    const float* gn_g   = (const float*)d_in[16]; const float* gn_b = (const float*)d_in[17];
    const float* dw1    = (const float*)d_in[18]; const float* pw1 = (const float*)d_in[19];
    const float* dw2    = (const float*)d_in[20]; const float* pw2 = (const float*)d_in[21];
    const float* bn0g = (const float*)d_in[22], *bn0b = (const float*)d_in[23], *bn0m = (const float*)d_in[24], *bn0v = (const float*)d_in[25];
    const float* bn1g = (const float*)d_in[26], *bn1b = (const float*)d_in[27], *bn1m = (const float*)d_in[28], *bn1v = (const float*)d_in[29];
    const float* bn2g = (const float*)d_in[30], *bn2b = (const float*)d_in[31], *bn2m = (const float*)d_in[32], *bn2v = (const float*)d_in[33];

    float* out_main = (float*)d_out;                                // (B, C, N)
    float* attn_out = out_main + (size_t)MROWS * N_;                // (B, H, C, C)

    // ---- workspace layout
    char* ws = (char*)d_ws;
    size_t off = 0;
    auto alloc = [&](size_t bytes) -> void* {
        off = (off + 255) & ~(size_t)255;
        void* p = ws + off; off += bytes; return p;
    };
    const size_t NN = (size_t)N_ * N_;
    bf16_t* Wq = (bf16_t*)alloc(NN * 2), *Wk = (bf16_t*)alloc(NN * 2);
    bf16_t* Wv = (bf16_t*)alloc(NN * 2), *Wp = (bf16_t*)alloc(NN * 2);
    bf16_t* Xq = (bf16_t*)alloc((size_t)MROWS * N_ * 2);
    bf16_t* Xk = (bf16_t*)alloc((size_t)MROWS * N_ * 2);
    bf16_t* Xv = (bf16_t*)alloc((size_t)MROWS * N_ * 2);
    float*  qf = (float*)alloc((size_t)MROWS * N_ * 4);
    float*  kf = (float*)alloc((size_t)MROWS * N_ * 4);
    float*  vf = (float*)alloc((size_t)MROWS * N_ * 4);
    float*  dwbuf = (float*)alloc((size_t)B_ * 234 * HW * 4);
    float*  pwbuf = (float*)alloc((size_t)B_ * 234 * HW * 4);
    float*  gst   = (float*)alloc((size_t)B_ * 2 * 2 * 4);
    bf16_t* qh  = (bf16_t*)alloc((size_t)B_ * HEADS * CP * D_ * 2);
    bf16_t* kh  = (bf16_t*)alloc((size_t)B_ * HEADS * CP * D_ * 2);
    bf16_t* vhT = (bf16_t*)alloc((size_t)B_ * HEADS * D_ * CP * 2);
    float*  scores = (float*)alloc((size_t)B_ * HEADS * CP * CP * 4);
    bf16_t* attnbf = (bf16_t*)alloc((size_t)B_ * HEADS * CP * CP * 2);
    float*  outpre = (float*)alloc((size_t)MROWS * N_ * 4);
    bf16_t* Xp     = (bf16_t*)alloc((size_t)MROWS * N_ * 2);

    // ---- 1. weights to bf16
    {
        int n = (int)NN, g = (n + 255) / 256;
        k_cast_bf16<<<g, 256, 0, stream>>>(wq_f, Wq, n);
        k_cast_bf16<<<g, 256, 0, stream>>>(wk_f, Wk, n);
        k_cast_bf16<<<g, 256, 0, stream>>>(wv_f, Wv, n);
        k_cast_bf16<<<g, 256, 0, stream>>>(wp_f, Wp, n);
    }
    // ---- 2. LayerNorm x3 -> bf16
    k_ln3<<<MROWS, 256, 0, stream>>>(x, ln_qg, ln_qb, ln_kg, ln_kb, ln_vg, ln_vb, Xq, Xk, Xv);

    // ---- 3. Q/K/V projections: (9984x1024) @ (1024x1024)^T
    dim3 gqkv(MROWS / 128, N_ / 128, 1);
    k_gemm_bf16<<<gqkv, 256, 0, stream>>>(Xq, Wq, qf, N_, N_, MROWS, 0, 0, 0, nullptr);
    k_gemm_bf16<<<gqkv, 256, 0, stream>>>(Xk, Wk, kf, N_, N_, MROWS, 0, 0, 0, nullptr);
    k_gemm_bf16<<<gqkv, 256, 0, stream>>>(Xv, Wv, vf, N_, N_, MROWS, 0, 0, 0, nullptr);

    // ---- 4. aggregator per tensor
    const long nconv = (long)B_ * 234 * HW;
    const int  gconv = (int)((nconv + 255) / 256);
    const long nfin  = (long)B_ * C_ * HW;
    const int  gfin  = (int)((nfin + 255) / 256);
    struct Cfg { const float* t; bf16_t* std; bf16_t* tr; } cfg[3] = {
        { qf, qh, nullptr }, { kf, kh, nullptr }, { vf, nullptr, vhT } };
    for (int i = 0; i < 3; ++i) {
        k_dw<<<gconv, 256, 0, stream>>>(cfg[i].t, dw1, dw2, dw0, dwbuf);
        k_pw<<<gconv, 256, 0, stream>>>(dwbuf, pw1, pw2, pw0, pwbuf);
        k_gnstats<<<B_ * 2, 256, 0, stream>>>(pwbuf, gst);
        k_finalize<<<gfin, 256, 0, stream>>>(cfg[i].t, pwbuf, gst,
            bn0g, bn0b, bn0m, bn0v, bn1g, bn1b, bn1m, bn1v, bn2g, bn2b, bn2m, bn2v,
            gn_g, gn_b, cfg[i].std, cfg[i].tr);
    }

    // ---- 5. scores = qh @ kh^T  (batched 384x384x128, pad computed/ignored)
    dim3 gsc(CP / 128, CP / 128, B_ * HEADS);
    k_gemm_bf16<<<gsc, 256, 0, stream>>>(qh, kh, scores, D_, CP, CP,
        (long)CP * D_, (long)CP * D_, (long)CP * CP, nullptr);

    // ---- 6. gelu * scale + softmax; emit attn (f32 output) + attn bf16 (zero-padded)
    dim3 gsm(CP, B_ * HEADS);
    k_softmax<<<gsm, 128, 0, stream>>>(scores, attn_out, attnbf);

    // ---- 7. out = attn @ vh  -> scattered so outpre is the torch "raw reshape" (9984,1024)
    dim3 gav(CP / 128, D_ / 128, B_ * HEADS);
    k_gemm_bf16<<<gav, 256, 0, stream>>>(attnbf, vhT, outpre, CP, D_, C_,
        (long)CP * CP, (long)D_ * CP, (long)C_ * D_, nullptr);

    // ---- 8. cast to bf16, proj GEMM (+bias) into d_out, then L2-normalize rows
    {
        int n = MROWS * N_, g = (n + 255) / 256;
        k_cast_bf16<<<g, 256, 0, stream>>>(outpre, Xp, n);
    }
    k_gemm_bf16<<<gqkv, 256, 0, stream>>>(Xp, Wp, out_main, N_, N_, MROWS, 0, 0, 0, b_proj);
    k_rownorm<<<MROWS, 256, 0, stream>>>(out_main);
}
